// MultiHeadAttention_29222957482206
// MI455X (gfx1250) — matrix operand order repaired and checked by reading, not compiled, not run on hardware
//
#include <hip/hip_runtime.h>

// ---------------------------------------------------------------------------
// MultiHeadAttention for MI455X (gfx1250, wave32, WMMA bf16 16x16x32)
// B=2, T=2048, C=1024, H=16, D=64
// ---------------------------------------------------------------------------

#define Bn   2
#define Tn   2048
#define Cn   1024
#define Hn   16
#define Dn   64
#define Mn   (Bn * Tn)      // 4096 tokens
#define Kn   Cn             // 1024 reduction dim for projections

typedef __bf16 v16bf __attribute__((ext_vector_type(16)));
typedef __bf16 v8bf  __attribute__((ext_vector_type(8)));
typedef __bf16 v4bf  __attribute__((ext_vector_type(4)));
typedef float  v8f   __attribute__((ext_vector_type(8)));

static __device__ __forceinline__ __bf16 f2bf(float f) {
    union { float f; unsigned u; } c; c.f = f;
    unsigned u = c.u + 0x7FFFu + ((c.u >> 16) & 1u);   // round-to-nearest-even
    union { unsigned short s; __bf16 b; } o; o.s = (unsigned short)(u >> 16);
    return o.b;
}

static __device__ __forceinline__ v16bf combine16(v8bf lo, v8bf hi) {
    v16bf r;
#pragma unroll
    for (int i = 0; i < 8; ++i) { r[i] = lo[i]; r[i + 8] = hi[i]; }
    return r;
}

#define WMMA_BF16(Afrag, Bfrag, Cfrag) \
    __builtin_amdgcn_wmma_f32_16x16x32_bf16(false, (Afrag), false, (Bfrag), (short)0, (Cfrag), false, false)

// Async global -> LDS 16B copy (per lane). ldsoff = wave-relative LDS byte
// address (low 32 bits of the flat shared pointer), gaddr = 64-bit global.
static __device__ __forceinline__ void async_copy_b128(unsigned ldsoff, const void* gaddr) {
    asm volatile("global_load_async_to_lds_b128 %0, %1, off"
                 :: "v"(ldsoff), "v"((unsigned long long)(size_t)gaddr) : "memory");
}
static __device__ __forceinline__ void wait_async0() {
    asm volatile("s_wait_asynccnt 0x0" ::: "memory");
}

// ---------------------------------------------------------------------------
// x (f32) -> bf16, 4 elements / thread
// ---------------------------------------------------------------------------
__global__ __launch_bounds__(256) void k_cvt_bf16(const float* __restrict__ in,
                                                  __bf16* __restrict__ out, int n4) {
    int i = blockIdx.x * blockDim.x + threadIdx.x;
    if (i >= n4) return;
    float4 v = ((const float4*)in)[i];
    v4bf o; o[0] = f2bf(v.x); o[1] = f2bf(v.y); o[2] = f2bf(v.z); o[3] = f2bf(v.w);
    ((v4bf*)out)[i] = o;
}

// ---------------------------------------------------------------------------
// W [K=in][N=out] f32  ->  Wt [N][K] bf16   (LDS-tiled 32x32 transpose)
// ---------------------------------------------------------------------------
__global__ __launch_bounds__(256) void k_transpose_bf16(const float* __restrict__ W,
                                                        __bf16* __restrict__ Wt) {
    __shared__ float tile[32][33];
    int tx = threadIdx.x, ty = threadIdx.y;               // (32, 8)
    int bx = blockIdx.x * 32, by = blockIdx.y * 32;
#pragma unroll
    for (int i = 0; i < 32; i += 8)
        tile[ty + i][tx] = W[(by + ty + i) * Cn + bx + tx];
    __syncthreads();
#pragma unroll
    for (int i = 0; i < 32; i += 8)
        Wt[(bx + ty + i) * Cn + by + tx] = f2bf(tile[tx][ty + i]);
}

// ---------------------------------------------------------------------------
// GEMM: out[M=4096, N=1024] = A[M,K=1024] @ Bt[N,K]^T   (bf16 in, f32 acc)
// block = 256 thr = 8 waves, block tile 128(M) x 64(N); wave tile 16x64.
// B tile staged in LDS via global_load_async_to_lds_b128, double-buffered;
// rows padded to 40 bf16 so B-frag ds_load_b128 is bank-conflict free.
// A frags double-buffered in registers so prefetch issues before the barrier.
// MODE 0: write bf16 [B,H,T,D]      (Q, K)
// MODE 1: write bf16 [B,H,D,T]      (V transposed, packed 16B stores)
// MODE 2: write f32  [M,C] + bias   (final output)
// ---------------------------------------------------------------------------
template <int MODE>
__global__ __launch_bounds__(256) void k_gemm_wmma(const __bf16* __restrict__ A,
                                                   const __bf16* __restrict__ Bt,
                                                   __bf16* __restrict__ outb,
                                                   float* __restrict__ outf,
                                                   const float* __restrict__ bias) {
    __shared__ __bf16 Bs[2][64 * 40];     // 2 x 5KB double buffer

    const int tid  = threadIdx.x;
    const int lane = tid & 31;
    const int wave = tid >> 5;
    const int nl   = lane & 15;           // row/col within 16-tile
    const int half = lane >> 4;           // K-half / M-half selector
    const int mBase = blockIdx.y * 128 + wave * 16;
    const int nBase = blockIdx.x * 64;
    const int srow   = tid >> 2;          // 0..63 : B-tile row this thread stages
    const int schunk = tid & 3;           // 0..3  : 8-element chunk within 32 K

    v8f acc[4];
#pragma unroll
    for (int t = 0; t < 4; ++t) acc[t] = (v8f)0.0f;

    const __bf16* arow = A + (size_t)(mBase + nl) * Kn;
    const __bf16* bsrc = Bt + (size_t)(nBase + srow) * Kn + schunk * 8;
    const unsigned ldsrow = (unsigned)(size_t)&Bs[0][srow * 40 + schunk * 8];
    const unsigned ldsbufstride = (unsigned)(64 * 40 * sizeof(__bf16));

    const int NT = Kn / 32;
    // ---- prologue: stage tile 0, load A frag 0 ----
    async_copy_b128(ldsrow, bsrc);
    v8bf alo = *(const v8bf*)(arow + half * 8);
    v8bf ahi = *(const v8bf*)(arow + 16 + half * 8);
    wait_async0();
    __syncthreads();
    v16bf afrag = combine16(alo, ahi);

    for (int ki = 0; ki < NT; ++ki) {
        const int buf = ki & 1;
        v8bf nlo{}, nhi{};
        if (ki + 1 < NT) {
            // prefetch tile ki+1 (async -> LDS) and next A frag (global -> regs)
            async_copy_b128(ldsrow + (buf ^ 1) * ldsbufstride, bsrc + (ki + 1) * 32);
            nlo = *(const v8bf*)(arow + (ki + 1) * 32 + half * 8);
            nhi = *(const v8bf*)(arow + (ki + 1) * 32 + 16 + half * 8);
        }
#pragma unroll
        for (int t = 0; t < 4; ++t) {
            // B-frag: lane n=nl holds K = half*8 + e (e<8), 16 + half*8 + (e-8) (e>=8)
            const __bf16* bb = &Bs[buf][(t * 16 + nl) * 40 + half * 8];
            v16bf bfrag = combine16(*(const v8bf*)bb, *(const v8bf*)(bb + 16));
            acc[t] = WMMA_BF16(afrag, bfrag, acc[t]);
        }
        if (ki + 1 < NT) {
            afrag = combine16(nlo, nhi);
            wait_async0();        // tile ki+1 written to LDS by this wave
            __syncthreads();      // visible to all waves; buffer `buf` now free
        }
    }

    // C/D layout: VGPR r -> M = r + 8*half, N = nl
#pragma unroll
    for (int t = 0; t < 4; ++t) {
        const int channel = nBase + t * 16 + nl;
        if (MODE == 2) {
            const float bv = bias[channel];
#pragma unroll
            for (int r = 0; r < 8; ++r) {
                const int token = mBase + r + 8 * half;
                outf[(size_t)token * Cn + channel] = acc[t][r] + bv;
            }
        } else if (MODE == 0) {
            const int h = channel >> 6, d = channel & 63;
#pragma unroll
            for (int r = 0; r < 8; ++r) {
                const int token = mBase + r + 8 * half;
                const int b = token >> 11, tt = token & (Tn - 1);
                outb[(((size_t)(b * Hn + h)) * Tn + tt) * Dn + d] = f2bf(acc[t][r]);
            }
        } else { // MODE 1: V transposed [B,H,D,T]; 8 consecutive tokens -> 16B store
            const int h = channel >> 6, d = channel & 63;
            const int t0 = mBase + 8 * half;
            const int b = t0 >> 11, tt0 = t0 & (Tn - 1);
            union { unsigned short s[8]; uint4 v; } pk;
#pragma unroll
            for (int r = 0; r < 8; ++r) {
                union { __bf16 b; unsigned short s; } cv; cv.b = f2bf(acc[t][r]);
                pk.s[r] = cv.s;
            }
            *(uint4*)(outb + (((size_t)(b * Hn + h)) * Dn + d) * Tn + tt0) = pk.v;
        }
    }
}

// ---------------------------------------------------------------------------
// Flash attention: 1 wave = one 16-query tile of one (b,h); 64-key blocks.
// Qb,Kb: bf16 [B,H,T,D]; Vt: bf16 [B,H,D,T]; Ob: bf16 [B,T,C]
// ---------------------------------------------------------------------------
__global__ __launch_bounds__(128) void k_attn_wmma(const __bf16* __restrict__ Qb,
                                                   const __bf16* __restrict__ Kb,
                                                   const __bf16* __restrict__ Vt,
                                                   __bf16* __restrict__ Ob) {
    __shared__ __bf16 Plds[4][16 * 64];               // 2KB per wave P staging

    const int lane = threadIdx.x & 31;
    const int wave = threadIdx.x >> 5;
    const int nl   = lane & 15;
    const int half = lane >> 4;
    const int bh = blockIdx.x;                        // 0..31  (b*H + h)
    const int q0 = (blockIdx.y * 4 + wave) * 16;

    // Q A-frags for the two K-steps over D=64
    v16bf qf[2];
    const __bf16* qrow = Qb + ((size_t)bh * Tn + q0 + nl) * Dn;
#pragma unroll
    for (int ks = 0; ks < 2; ++ks)
        qf[ks] = combine16(*(const v8bf*)(qrow + ks * 32 + half * 8),
                           *(const v8bf*)(qrow + ks * 32 + 16 + half * 8));

    v8f o[4];
#pragma unroll
    for (int t = 0; t < 4; ++t) o[t] = (v8f)0.0f;
    float mrun[8], lrun[8];
#pragma unroll
    for (int r = 0; r < 8; ++r) { mrun[r] = -1.0e30f; lrun[r] = 0.0f; }

    __bf16* pw = &Plds[wave][0];
    const int nblocks = (q0 + 16 + 63) >> 6;

    for (int kb = 0; kb < nblocks; ++kb) {
        const int k0 = kb * 64;
        // ---- S = (Q K^T) * 1/sqrt(D), causal mask ----
        v8f s[4];
#pragma unroll
        for (int t = 0; t < 4; ++t) {
            v8f sc = (v8f)0.0f;
            const __bf16* krow = Kb + ((size_t)bh * Tn + k0 + t * 16 + nl) * Dn;
#pragma unroll
            for (int ks = 0; ks < 2; ++ks) {
                v16bf kf = combine16(*(const v8bf*)(krow + ks * 32 + half * 8),
                                     *(const v8bf*)(krow + ks * 32 + 16 + half * 8));
                sc = WMMA_BF16(qf[ks], kf, sc);
            }
            const int key = k0 + t * 16 + nl;
#pragma unroll
            for (int r = 0; r < 8; ++r) {
                const int q = q0 + r + 8 * half;
                s[t][r] = (key <= q) ? sc[r] * 0.125f : -1.0e30f;
            }
        }
        // ---- online softmax: block row-max over 64 keys ----
        float bm[8];
#pragma unroll
        for (int r = 0; r < 8; ++r)
            bm[r] = fmaxf(fmaxf(s[0][r], s[1][r]), fmaxf(s[2][r], s[3][r]));
#pragma unroll
        for (int r = 0; r < 8; ++r)
#pragma unroll
            for (int off = 8; off > 0; off >>= 1)
                bm[r] = fmaxf(bm[r], __shfl_xor(bm[r], off));

        float corr[8], psum[8];
#pragma unroll
        for (int r = 0; r < 8; ++r) {
            const float mnew = fmaxf(mrun[r], bm[r]);
            corr[r] = __expf(mrun[r] - mnew);
            mrun[r] = mnew;
            psum[r] = 0.0f;
        }
        // ---- P = exp(S - m), stage to LDS for A-frag relayout ----
#pragma unroll
        for (int t = 0; t < 4; ++t)
#pragma unroll
            for (int r = 0; r < 8; ++r) {
                const float p = __expf(s[t][r] - mrun[r]);
                psum[r] += p;
                pw[(r + 8 * half) * 64 + t * 16 + nl] = f2bf(p);
            }
#pragma unroll
        for (int r = 0; r < 8; ++r)
#pragma unroll
            for (int off = 8; off > 0; off >>= 1)
                psum[r] += __shfl_xor(psum[r], off);
#pragma unroll
        for (int r = 0; r < 8; ++r) lrun[r] = lrun[r] * corr[r] + psum[r];
#pragma unroll
        for (int t = 0; t < 4; ++t)
#pragma unroll
            for (int r = 0; r < 8; ++r) o[t][r] *= corr[r];

        // same-wave LDS write -> read ordering
        asm volatile("s_wait_dscnt 0" ::: "memory");

        // ---- O += P @ V ----
#pragma unroll
        for (int ks = 0; ks < 2; ++ks) {
            const __bf16* pr = pw + nl * 64 + ks * 32 + half * 8;
            v16bf pa = combine16(*(const v8bf*)pr, *(const v8bf*)(pr + 16));
#pragma unroll
            for (int t = 0; t < 4; ++t) {
                const __bf16* vrow = Vt + ((size_t)bh * Dn + t * 16 + nl) * Tn
                                        + k0 + ks * 32 + half * 8;
                v16bf vf = combine16(*(const v8bf*)vrow, *(const v8bf*)(vrow + 16));
                o[t] = WMMA_BF16(pa, vf, o[t]);
            }
        }
    }

    // ---- epilogue: O / l, write bf16 [B,T,C] ----
    const int h = bh & (Hn - 1), b = bh >> 4;
#pragma unroll
    for (int t = 0; t < 4; ++t)
#pragma unroll
        for (int r = 0; r < 8; ++r) {
            const int q = q0 + r + 8 * half;
            Ob[((size_t)(b * Tn + q)) * Cn + h * Dn + t * 16 + nl] = f2bf(o[t][r] / lrun[r]);
        }
}

// ---------------------------------------------------------------------------
// Host-side launch
// ---------------------------------------------------------------------------
extern "C" void kernel_launch(void* const* d_in, const int* in_sizes, int n_in,
                              void* d_out, int out_size, void* d_ws, size_t ws_size,
                              hipStream_t stream) {
    const float* x   = (const float*)d_in[0];
    const float* Wq  = (const float*)d_in[1];
    const float* Wk  = (const float*)d_in[2];
    const float* Wv  = (const float*)d_in[3];
    const float* Wo  = (const float*)d_in[4];
    const float* bo  = (const float*)d_in[5];
    float* out = (float*)d_out;

    const size_t MiB = 1048576;
    char* ws = (char*)d_ws;
    __bf16* xb  = (__bf16*)(ws);               //  8 MiB  [M,K]
    __bf16* wtq = (__bf16*)(ws +  8 * MiB);    //  2 MiB  [N,K]
    __bf16* wtk = (__bf16*)(ws + 10 * MiB);
    __bf16* wtv = (__bf16*)(ws + 12 * MiB);
    __bf16* wto = (__bf16*)(ws + 14 * MiB);
    __bf16* qb  = (__bf16*)(ws + 16 * MiB);    //  8 MiB  [B,H,T,D]
    __bf16* kb  = (__bf16*)(ws + 24 * MiB);    //  8 MiB  [B,H,T,D]
    __bf16* vt  = (__bf16*)(ws + 32 * MiB);    //  8 MiB  [B,H,D,T]
    __bf16* ob  = (__bf16*)(ws + 40 * MiB);    //  8 MiB  [B,T,C]

    // x -> bf16
    k_cvt_bf16<<<(Mn * Kn / 4 + 255) / 256, 256, 0, stream>>>(x, xb, Mn * Kn / 4);
    // weights -> transposed bf16 [out, in]
    dim3 tb(32, 8), tg(Cn / 32, Cn / 32);
    k_transpose_bf16<<<tg, tb, 0, stream>>>(Wq, wtq);
    k_transpose_bf16<<<tg, tb, 0, stream>>>(Wk, wtk);
    k_transpose_bf16<<<tg, tb, 0, stream>>>(Wv, wtv);
    k_transpose_bf16<<<tg, tb, 0, stream>>>(Wo, wto);

    // Q,K,V projections (WMMA + async LDS staging)
    dim3 gg(Cn / 64, Mn / 128);
    k_gemm_wmma<0><<<gg, 256, 0, stream>>>(xb, wtq, qb, nullptr, nullptr);
    k_gemm_wmma<0><<<gg, 256, 0, stream>>>(xb, wtk, kb, nullptr, nullptr);
    k_gemm_wmma<1><<<gg, 256, 0, stream>>>(xb, wtv, vt, nullptr, nullptr);

    // flash attention (WMMA): 32 (b,h) x 32 blocks of 4 q-tiles
    k_attn_wmma<<<dim3(Bn * Hn, Tn / 16 / 4), 128, 0, stream>>>(qb, kb, vt, ob);

    // output projection + bias -> f32 d_out
    k_gemm_wmma<2><<<gg, 256, 0, stream>>>(ob, wto, nullptr, out, bo);
}
